// sBRU_28664611733941
// MI455X (gfx1250) — matrix operand order repaired and checked by reading, not compiled, not run on hardware
//
#include <hip/hip_runtime.h>
#include <hip/hip_bf16.h>
#include <math.h>

// ---------------------------------------------------------------------------
// sBRU (2 layers): Wx = LN(x @ W^T); h_t = softplus(Wx_t + h_{t-1} @ V^T)
// B=64, T=1024, F=80, H=512.
//   * bf16 WMMA (v_wmma_f32_16x16x32_bf16) for FF GEMM+LN and the recurrence.
//   * Recurrence: 16 workgroups, each owns 32 output columns, V-slice pinned
//     in LDS for all 1024 steps; h staged via async-to-LDS; grid barrier/step.
// ---------------------------------------------------------------------------

typedef __bf16 bf16_t;
typedef __attribute__((ext_vector_type(16))) __bf16 v16bf;
typedef __attribute__((ext_vector_type(8)))  float  v8f;
typedef __attribute__((ext_vector_type(4)))  int    v4i;

#define WAVE    32
#define HID     512
#define LPAD    8                    // LDS row padding (bf16 elems = 16B)

// ---- CDNA5 async global->LDS copy (ASYNCcnt-tracked, bypasses VGPRs) ------
#if defined(__gfx1250__) && __has_builtin(__builtin_amdgcn_global_load_async_to_lds_b128)
#define HAVE_ASYNC_LDS 1
typedef __attribute__((address_space(1))) v4i* gv4i_ptr;
typedef __attribute__((address_space(3))) v4i* lv4i_ptr;
#else
#define HAVE_ASYNC_LDS 0
#endif

__device__ __forceinline__ void copy16B_to_lds(const void* gsrc, void* ldst) {
#if HAVE_ASYNC_LDS
    __builtin_amdgcn_global_load_async_to_lds_b128(
        (gv4i_ptr)gsrc, (lv4i_ptr)ldst, 0, 0);
#else
    *(uint4*)ldst = *(const uint4*)gsrc;
#endif
}

__device__ __forceinline__ void async_wait0() {
#if HAVE_ASYNC_LDS
 #if __has_builtin(__builtin_amdgcn_s_wait_asynccnt)
    __builtin_amdgcn_s_wait_asynccnt(0);
 #else
    asm volatile("s_wait_asynccnt 0x0" ::: "memory");
 #endif
#endif
}

__device__ __forceinline__ float softplus_f(float x) {
    // numerically stable: max(x,0) + log1p(exp(-|x|))
    return fmaxf(x, 0.0f) + log1pf(__expf(-fabsf(x)));
}

// A fragment: 16x32 (MxK) bf16. lane&15 selects row; lanes<16 hold K 0-7/16-23,
// lanes>=16 hold K 8-15/24-31 (per CDNA5 ISA 16-bit A layout).
__device__ __forceinline__ v16bf frag_a(const bf16_t* __restrict__ buf,
                                        int rowStride, int m0, int kbase) {
    const int lane = threadIdx.x & (WAVE - 1);
    const int row  = m0 + (lane & 15);
    const int ksel = (lane < 16) ? 0 : 8;
    const bf16_t* p = buf + row * rowStride + kbase + ksel;
    v16bf a;
#pragma unroll
    for (int i = 0; i < 8; ++i) { a[i] = p[i]; a[8 + i] = p[16 + i]; }
    return a;
}

// B fragment: 32x16 (KxN) bf16 where B[k][n] = buf[n0+n][k] (B = slice^T).
// lane&15 selects n; lanes<16 hold K 0-7/16-23, lanes>=16 hold K 8-15/24-31.
__device__ __forceinline__ v16bf frag_b(const bf16_t* __restrict__ buf,
                                        int rowStride, int n0, int kbase) {
    const int lane  = threadIdx.x & (WAVE - 1);
    const int nrow  = n0 + (lane & 15);
    const int khalf = (lane < 16) ? 0 : 8;
    const bf16_t* p = buf + nrow * rowStride + kbase + khalf;
    v16bf b;
#pragma unroll
    for (int i = 0; i < 8; ++i) { b[i] = p[i]; b[8 + i] = p[16 + i]; }
    return b;
}

// fp32 -> bf16 convert with K padding (pad zeros so WMMA K-loop is exact)
__global__ __launch_bounds__(256)
void cvt_bf16_pad_kernel(const float* __restrict__ src, bf16_t* __restrict__ dst,
                         int rows, int K, int Kp) {
    int idx = blockIdx.x * 256 + threadIdx.x;
    int total = rows * Kp;
    if (idx >= total) return;
    int r = idx / Kp, k = idx - r * Kp;
    float v = (k < K) ? src[(size_t)r * K + k] : 0.0f;
    dst[idx] = (bf16_t)v;
}

// ---------------------------------------------------------------------------
// Feedforward GEMM + LayerNorm.  One block = 16 M-rows x all 512 N.
// X: fp32 [M,K]; Wb: bf16 [512,Kp] (row h = W[h,:] padded); Y: fp32 [M,512]
// ---------------------------------------------------------------------------
__global__ __launch_bounds__(256)
void ff_ln_kernel(const float* __restrict__ X, const bf16_t* __restrict__ Wb,
                  const float* __restrict__ gam, const float* __restrict__ bet,
                  float* __restrict__ Y, int K, int Kp) {
    __shared__ bf16_t lx[16 * (HID + LPAD)];   // staged X tile (bf16)
    __shared__ float  lo[16 * HID];            // pre-LN output tile
    __shared__ float  lmu[16], lrs[16];

    const int m0   = blockIdx.x * 16;
    const int tid  = threadIdx.x;
    const int wave = tid / WAVE;
    const int lane = tid & (WAVE - 1);
    const int lxs  = Kp + LPAD;                // padded LDS row stride

    // stage + convert X tile (zero-pad K..Kp)
    for (int idx = tid; idx < 16 * Kp; idx += 256) {
        int r = idx / Kp, k = idx - r * Kp;
        float v = (k < K) ? X[(size_t)(m0 + r) * K + k] : 0.0f;
        lx[r * lxs + k] = (bf16_t)v;
    }
    __syncthreads();

    const v8f vzero = {0.f, 0.f, 0.f, 0.f, 0.f, 0.f, 0.f, 0.f};
    v8f acc[4];
#pragma unroll
    for (int i = 0; i < 4; ++i) acc[i] = vzero;

    for (int kt = 0; kt < Kp; kt += 32) {
        v16bf a = frag_a(lx, lxs, 0, kt);
        // load all 4 B fragments into distinct registers first so the
        // scheduler can overlap VMEM with the WMMA pipe (partial loadcnt waits)
        v16bf bfr[4];
#pragma unroll
        for (int i = 0; i < 4; ++i)
            bfr[i] = frag_b(Wb, Kp, (wave + 8 * i) * 16, kt);
#pragma unroll
        for (int i = 0; i < 4; ++i)
            acc[i] = __builtin_amdgcn_wmma_f32_16x16x32_bf16(
                         false, a, false, bfr[i], (short)0, acc[i], false, false);
    }

    // spill C tiles to LDS (C layout: vgpr r -> row r (+8 for upper lanes))
    {
        const int rbase = (lane < 16) ? 0 : 8;
        const int col   = lane & 15;
#pragma unroll
        for (int i = 0; i < 4; ++i) {
            const int n0 = (wave + 8 * i) * 16;
#pragma unroll
            for (int r = 0; r < 8; ++r)
                lo[(rbase + r) * HID + n0 + col] = acc[i][r];
        }
    }
    __syncthreads();

    // LayerNorm: each wave reduces 2 rows with wave32 shuffles
#pragma unroll
    for (int rr = 0; rr < 2; ++rr) {
        const int row = wave * 2 + rr;
        float s = 0.f, ss = 0.f;
        for (int c = lane; c < HID; c += WAVE) {
            float v = lo[row * HID + c];
            s += v; ss += v * v;
        }
#pragma unroll
        for (int o = 16; o; o >>= 1) {
            s  += __shfl_xor(s,  o, WAVE);
            ss += __shfl_xor(ss, o, WAVE);
        }
        if (lane == 0) {
            float mu  = s * (1.0f / HID);
            float var = ss * (1.0f / HID) - mu * mu;
            lmu[row] = mu;
            lrs[row] = rsqrtf(var + 1e-5f);
        }
    }
    __syncthreads();

    for (int idx = tid; idx < 16 * HID; idx += 256) {
        int r = idx >> 9, c = idx & (HID - 1);
        float v = (lo[idx] - lmu[r]) * lrs[r] * gam[c] + bet[c];
        Y[(size_t)(m0 + r) * HID + c] = v;
    }
}

// ---------------------------------------------------------------------------
// Recurrence: 16 workgroups x 256 threads; workgroup owns 32 output columns.
// V-slice lives in LDS for the whole T loop; h_{t-1} staged per step via
// async-to-LDS; grid barrier via device-scope atomic counter.
// ---------------------------------------------------------------------------
#define REC_NW 16
#define LHS    (HID + LPAD)     // padded LDS stride

__global__ __launch_bounds__(256)
void rec_kernel(const float* __restrict__ Wx,    // [B,T,H] fp32 (post-LN)
                const bf16_t* __restrict__ Vb,   // [H,H] bf16, row j = V[j,:]
                float* __restrict__ Hout,        // [B,T,H] fp32
                bf16_t* __restrict__ hstate,     // [2][B*H] bf16, buf0 zeroed
                unsigned int* __restrict__ sync_cnt,
                int B, int T) {
    __shared__ bf16_t lV[32 * LHS];   // V rows j0..j0+31  (~33 KB)
    __shared__ bf16_t lH[64 * LHS];   // previous h, all 64 batch rows (~66 KB)

    const int tid  = threadIdx.x;
    const int wave = tid / WAVE;
    const int lane = tid & (WAVE - 1);
    const int j0   = blockIdx.x * 32;

    // pin this workgroup's V slice in LDS for all timesteps (16B chunks)
    for (int ch = tid; ch < 32 * (HID / 8); ch += 256) {
        int r = ch >> 6, c0 = (ch & 63) * 8;
        copy16B_to_lds(&Vb[(size_t)(j0 + r) * HID + c0], &lV[r * LHS + c0]);
    }
    async_wait0();

    const int mt    = wave & 3;              // 4 M-tiles over B=64
    const int nt    = wave >> 2;             // 2 N-tiles over 32 cols
    const int rbase = (lane < 16) ? 0 : 8;
    const int col   = lane & 15;
    const int jcol  = j0 + nt * 16 + col;    // this lane's hidden index
    const v8f vzero = {0.f, 0.f, 0.f, 0.f, 0.f, 0.f, 0.f, 0.f};

    for (int t = 0; t < T; ++t) {
        const bf16_t* hprev = hstate + (size_t)(t & 1) * (64 * HID);
        bf16_t*       hnext = hstate + (size_t)((t + 1) & 1) * (64 * HID);

        // warm L2/L0 for this step's strided Wx epilogue loads
#pragma unroll
        for (int r = 0; r < 8; ++r) {
            const int brow = mt * 16 + rbase + r;
            __builtin_prefetch(&Wx[((size_t)brow * T + t) * HID + jcol], 0, 0);
        }

        // stage h_{t-1} (64x512 bf16 = 64KB) into LDS, 16B chunks
        for (int ch = tid; ch < 64 * (HID / 8); ch += 256) {
            int r = ch >> 6, c0 = (ch & 63) * 8;
            copy16B_to_lds(&hprev[(size_t)r * HID + c0], &lH[r * LHS + c0]);
        }
        async_wait0();
        __syncthreads();

        // one 16x16 tile per wave: C = h_prev @ V^T (K = 512),
        // two independent accumulators to break the dependent WMMA chain
        v8f c0a = vzero, c1a = vzero;
        for (int kt = 0; kt < HID; kt += 64) {
            v16bf a0 = frag_a(lH, LHS, mt * 16, kt);
            v16bf b0 = frag_b(lV, LHS, nt * 16, kt);
            v16bf a1 = frag_a(lH, LHS, mt * 16, kt + 32);
            v16bf b1 = frag_b(lV, LHS, nt * 16, kt + 32);
            c0a = __builtin_amdgcn_wmma_f32_16x16x32_bf16(
                      false, a0, false, b0, (short)0, c0a, false, false);
            c1a = __builtin_amdgcn_wmma_f32_16x16x32_bf16(
                      false, a1, false, b1, (short)0, c1a, false, false);
        }

        // epilogue: softplus(c + Wx[:,t,slice]) -> output + bf16 feedback
#pragma unroll
        for (int r = 0; r < 8; ++r) {
            const int brow = mt * 16 + rbase + r;       // batch index
            const size_t o = ((size_t)brow * T + t) * HID + jcol;
            float v = softplus_f(c0a[r] + c1a[r] + Wx[o]);
            Hout[o] = v;
            hnext[(size_t)brow * HID + jcol] = (bf16_t)v;
        }

        // device-wide step barrier
        __threadfence();
        __syncthreads();
        if (tid == 0) {
            const unsigned int target = (unsigned int)gridDim.x * (t + 1);
            __hip_atomic_fetch_add(sync_cnt, 1u, __ATOMIC_ACQ_REL,
                                   __HIP_MEMORY_SCOPE_AGENT);
            while (__hip_atomic_load(sync_cnt, __ATOMIC_ACQUIRE,
                                     __HIP_MEMORY_SCOPE_AGENT) < target)
                __builtin_amdgcn_s_sleep(1);
        }
        __syncthreads();
    }
}

// ---------------------------------------------------------------------------
// Host side
// ---------------------------------------------------------------------------
extern "C" void kernel_launch(void* const* d_in, const int* in_sizes, int n_in,
                              void* d_out, int out_size, void* d_ws, size_t ws_size,
                              hipStream_t stream) {
    (void)in_sizes; (void)n_in; (void)out_size; (void)ws_size;

    const float* x  = (const float*)d_in[0];
    const float* W0 = (const float*)d_in[1];
    const float* V0 = (const float*)d_in[2];
    const float* g0 = (const float*)d_in[3];
    const float* b0 = (const float*)d_in[4];
    const float* W1 = (const float*)d_in[5];
    const float* V1 = (const float*)d_in[6];
    const float* g1 = (const float*)d_in[7];
    const float* b1 = (const float*)d_in[8];
    float* out = (float*)d_out;

    const int B = 64, T = 1024, F = 80, H = HID;
    const int M = B * T;

    // workspace layout: Wx (128MB) | Wbf (512KB) | Vbf (512KB) | hstate | cnt
    char* ws = (char*)d_ws;
    size_t off = 0;
    float*  Wx  = (float*)(ws + off);  off += (size_t)M * H * sizeof(float);
    bf16_t* Wb  = (bf16_t*)(ws + off); off += (size_t)H * H * sizeof(bf16_t);
    bf16_t* Vb  = (bf16_t*)(ws + off); off += (size_t)H * H * sizeof(bf16_t);
    bf16_t* hst = (bf16_t*)(ws + off); off += 2 * (size_t)B * H * sizeof(bf16_t);
    unsigned int* cnt = (unsigned int*)(ws + off);

    // ---- layer 0 (K=80 padded to 96) ----
    {
        const int K = F, Kp = 96;
        cvt_bf16_pad_kernel<<<(H * Kp + 255) / 256, 256, 0, stream>>>(W0, Wb, H, K, Kp);
        ff_ln_kernel<<<M / 16, 256, 0, stream>>>(x, Wb, g0, b0, Wx, K, Kp);
        cvt_bf16_pad_kernel<<<(H * H + 255) / 256, 256, 0, stream>>>(V0, Vb, H, H, H);
        (void)hipMemsetAsync(hst, 0, (size_t)B * H * sizeof(bf16_t), stream);
        (void)hipMemsetAsync(cnt, 0, sizeof(unsigned int), stream);
        rec_kernel<<<REC_NW, 256, 0, stream>>>(Wx, Vb, out, hst, cnt, B, T);
    }
    // ---- layer 1 (K=512) ----
    {
        const int K = H, Kp = H;
        cvt_bf16_pad_kernel<<<(H * Kp + 255) / 256, 256, 0, stream>>>(W1, Wb, H, K, Kp);
        ff_ln_kernel<<<M / 16, 256, 0, stream>>>(out, Wb, g1, b1, Wx, K, Kp);
        cvt_bf16_pad_kernel<<<(H * H + 255) / 256, 256, 0, stream>>>(V1, Vb, H, H, H);
        (void)hipMemsetAsync(hst, 0, (size_t)B * H * sizeof(bf16_t), stream);
        (void)hipMemsetAsync(cnt, 0, sizeof(unsigned int), stream);
        rec_kernel<<<REC_NW, 256, 0, stream>>>(Wx, Vb, out, hst, cnt, B, T);
    }
}
